// SwarmNet_24008867184732
// MI455X (gfx1250) — compile-verified, matrix operand order checked by reading, not run on hardware
//
#include <hip/hip_runtime.h>

// CDNA5 / gfx1250: wave32, WMMA 16x16x32 f16 (f32 accumulate).
typedef __attribute__((ext_vector_type(16))) _Float16 v16h;
typedef __attribute__((ext_vector_type(8)))  float    v8f;

#define NT 256   // output timesteps (262 - 6)
#define NA 128   // agents
#define NH 32    // hidden width

// ---------------------------------------------------------------------------
// A-fragment (16x32, f16) from an LDS f32 tile, row-major [.. ][32].
// ISA 7.12.2: lane L holds row M=L%16; halves 0..7 = K kb..kb+7,
// halves 8..15 = K 16+kb..16+kb+7, where kb = (L/16)*8.
// ---------------------------------------------------------------------------
__device__ __forceinline__ v16h make_A(const float (*S)[NH], int rowbase, int lane) {
    int row = rowbase + (lane & 15);
    int kb  = (lane >> 4) << 3;
    v16h a;
#pragma unroll
    for (int j = 0; j < 8; ++j) {
        a[j]     = (_Float16)S[row][kb + j];
        a[8 + j] = (_Float16)S[row][16 + kb + j];
    }
    return a;
}

// B-fragment (32x16, f16) from global f32 weights W (KxN row-major, stride ldn),
// columns colbase..colbase+15, K = 0..31. Mirrors the A layout with N per lane.
__device__ __forceinline__ v16h make_B(const float* __restrict__ W, int ldn,
                                       int colbase, int lane) {
    int col = colbase + (lane & 15);
    int kb  = (lane >> 4) << 3;
    v16h b;
#pragma unroll
    for (int j = 0; j < 8; ++j) {
        b[j]     = (_Float16)W[(kb + j) * ldn + col];
        b[8 + j] = (_Float16)W[(16 + kb + j) * ldn + col];
    }
    return b;
}

// ---------------------------------------------------------------------------
// Kernel 1: temporal conv stack (4 -> 32 -> 64 -> 32, k=3, VALID), per agent,
// time-tiled in chunks of 64 outputs (+ halo) so LDS stays ~27 KB.
// grid = (4 chunks, 128 agents), block = 256.
// ---------------------------------------------------------------------------
__global__ void swarm_conv_kernel(const float* __restrict__ x,   // [262][128][4]
                                  const float* __restrict__ w1, const float* __restrict__ b1,
                                  const float* __restrict__ w2, const float* __restrict__ b2,
                                  const float* __restrict__ w3, const float* __restrict__ b3,
                                  float* __restrict__ ts)        // [256][128][32]
{
    __shared__ float xs[70][4];
    __shared__ float h1[68][32];
    __shared__ float h2[66][64];

    const int t0  = blockIdx.x * 64;   // 0,64,128,192
    const int a   = blockIdx.y;        // agent
    const int tid = threadIdx.x;

    for (int i = tid; i < 70 * 4; i += 256) {
        int u = i >> 2, f = i & 3;
        xs[u][f] = x[((t0 + u) * NA + a) * 4 + f];
    }
    __syncthreads();

    // conv1: 4 -> 32
    for (int i = tid; i < 68 * 32; i += 256) {
        int u = i >> 5, co = i & 31;
        float acc = b1[co];
#pragma unroll
        for (int ci = 0; ci < 4; ++ci)
#pragma unroll
            for (int k = 0; k < 3; ++k)
                acc = fmaf(w1[co * 12 + ci * 3 + k], xs[u + k][ci], acc);
        h1[u][co] = fmaxf(acc, 0.f);
    }
    __syncthreads();

    // conv2: 32 -> 64
    for (int i = tid; i < 66 * 64; i += 256) {
        int u = i >> 6, co = i & 63;
        float acc = b2[co];
        for (int ci = 0; ci < 32; ++ci)
#pragma unroll
            for (int k = 0; k < 3; ++k)
                acc = fmaf(w2[co * 96 + ci * 3 + k], h1[u + k][ci], acc);
        h2[u][co] = fmaxf(acc, 0.f);
    }
    __syncthreads();

    // conv3: 64 -> 32, write ts[t][a][c]
    for (int i = tid; i < 64 * 32; i += 256) {
        int u = i >> 5, co = i & 31;
        float acc = b3[co];
        for (int ci = 0; ci < 64; ++ci)
#pragma unroll
            for (int k = 0; k < 3; ++k)
                acc = fmaf(w3[co * 192 + ci * 3 + k], h2[u + k][ci], acc);
        ts[((t0 + u) * NA + a) * NH + co] = fmaxf(acc, 0.f);
    }
}

// ---------------------------------------------------------------------------
// Kernel 2: fused per-timestep graph network. One block per timestep t,
// 256 threads = 8 wave32s. All GEMMs via v_wmma_f32_16x16x32_f16.
// LDS aliasing: eout_s <- agg_pre <- nodes, ein_s <- agg.
// ---------------------------------------------------------------------------
__global__ void swarm_gnn_kernel(const float* __restrict__ ts_g,  // [256][128][32]
                                 const float* __restrict__ x,     // [262][128][4]
                                 const float* __restrict__ edge_w, const float* __restrict__ edge_b,
                                 const float* __restrict__ agg_w,  const float* __restrict__ agg_b,
                                 const float* __restrict__ node_w, const float* __restrict__ node_b,
                                 const float* __restrict__ dec_w,  const float* __restrict__ dec_b,
                                 float* __restrict__ out)          // [256][128][1][4]
{
    __shared__ float ts_s[NA][NH];    // ts[t]
    __shared__ float ein_s[NA][NH];   // e_in, later agg
    __shared__ float eout_s[NA][NH];  // e_out, later agg_pre, later nodes

    const int t    = blockIdx.x;
    const int tid  = threadIdx.x;
    const int lane = tid & 31;        // wave32
    const int wave = tid >> 5;        // 0..7 -> 16-row M-block

    for (int i = tid; i < NA * NH; i += 256)
        (&ts_s[0][0])[i] = ts_g[t * NA * NH + i];
    __syncthreads();

    const int ncol  = lane & 15;
    const int rbase = wave * 16 + ((lane >> 4) << 3);  // C/D row base (ISA 7.12.2)

    // ---- edge projections: [e_in | e_out] = ts @ [edge_w[:32] | edge_w[32:]] ----
    {
        v16h A = make_A(ts_s, wave * 16, lane);
#pragma unroll
        for (int nt = 0; nt < 4; ++nt) {
            const float* W = (nt < 2) ? edge_w : (edge_w + NH * NH);
            v16h B = make_B(W, NH, (nt & 1) * 16, lane);
            v8f  C = {};
            C = __builtin_amdgcn_wmma_f32_16x16x32_f16(false, A, false, B,
                                                       (short)0, C, false, false);
            float (*dst)[NH] = (nt < 2) ? ein_s : eout_s;
            int col = (nt & 1) * 16 + ncol;
#pragma unroll
            for (int r = 0; r < 8; ++r) dst[rbase + r][col] = C[r];
        }
    }
    __syncthreads();

    // ---- nonlinear all-pairs aggregation (the dominant cost, never materialized):
    // agg_pre[i][c] = sum_j relu(e_out[i][c] + e_in[j][c] + edge_b[c])
    {
        const int i  = tid >> 1;
        const int ch = (tid & 1) * 16;
        float eo[16], acc[16];
#pragma unroll
        for (int c = 0; c < 16; ++c) {
            eo[c]  = eout_s[i][ch + c] + edge_b[ch + c];
            acc[c] = 0.f;
        }
        for (int j = 0; j < NA; ++j)
#pragma unroll
            for (int c = 0; c < 16; ++c)
                acc[c] += fmaxf(eo[c] + ein_s[j][ch + c], 0.f);
        // overwrite e_out in place: this thread is the only reader of eout_s[i][ch..ch+15]
#pragma unroll
        for (int c = 0; c < 16; ++c) eout_s[i][ch + c] = acc[c];
    }
    __syncthreads();

    // ---- agg = relu(agg_pre @ agg_w + agg_b), writes into ein_s ----
    {
        v16h A = make_A(eout_s, wave * 16, lane);
#pragma unroll
        for (int nt = 0; nt < 2; ++nt) {
            v16h B = make_B(agg_w, NH, nt * 16, lane);
            v8f  C = {};
            C = __builtin_amdgcn_wmma_f32_16x16x32_f16(false, A, false, B,
                                                       (short)0, C, false, false);
            int   col  = nt * 16 + ncol;
            float bias = agg_b[col];
#pragma unroll
            for (int r = 0; r < 8; ++r)
                ein_s[rbase + r][col] = fmaxf(C[r] + bias, 0.f);
        }
    }
    __syncthreads();

    // ---- nodes = relu([ts, agg] @ node_w + node_b): K=64 as two K=32 WMMAs ----
    {
        v16h A1 = make_A(ts_s,  wave * 16, lane);   // node_w rows 0..31
        v16h A2 = make_A(ein_s, wave * 16, lane);   // node_w rows 32..63
#pragma unroll
        for (int nt = 0; nt < 2; ++nt) {
            v16h B1 = make_B(node_w,           NH, nt * 16, lane);
            v16h B2 = make_B(node_w + NH * NH, NH, nt * 16, lane);
            v8f  C  = {};
            C = __builtin_amdgcn_wmma_f32_16x16x32_f16(false, A1, false, B1,
                                                       (short)0, C, false, false);
            C = __builtin_amdgcn_wmma_f32_16x16x32_f16(false, A2, false, B2,
                                                       (short)0, C, false, false);
            int   col  = nt * 16 + ncol;
            float bias = node_b[col];
#pragma unroll
            for (int r = 0; r < 8; ++r)
                eout_s[rbase + r][col] = fmaxf(C[r] + bias, 0.f);  // nodes
        }
    }
    __syncthreads();

    // ---- decode + residual: out[t,a,:2] = x[t+6,a,:2] + nodes@dec_w + dec_b ----
    {
        const int a = tid >> 1, p = tid & 1;
        float acc = dec_b[p];
        for (int c = 0; c < NH; ++c)
            acc = fmaf(eout_s[a][c], dec_w[c * 2 + p], acc);
        const int xb = ((t + 6) * NA + a) * 4;
        const int ob = (t * NA + a) * 4;
        out[ob + p]     = x[xb + p] + acc;
        out[ob + 2 + p] = x[xb + 2 + p];   // passthrough features 2,3
    }
}

// ---------------------------------------------------------------------------
extern "C" void kernel_launch(void* const* d_in, const int* in_sizes, int n_in,
                              void* d_out, int out_size, void* d_ws, size_t ws_size,
                              hipStream_t stream) {
    const float* x      = (const float*)d_in[0];
    const float* w1     = (const float*)d_in[1];
    const float* b1     = (const float*)d_in[2];
    const float* w2     = (const float*)d_in[3];
    const float* b2     = (const float*)d_in[4];
    const float* w3     = (const float*)d_in[5];
    const float* b3     = (const float*)d_in[6];
    const float* edge_w = (const float*)d_in[7];
    const float* edge_b = (const float*)d_in[8];
    const float* agg_w  = (const float*)d_in[9];
    const float* agg_b  = (const float*)d_in[10];
    const float* node_w = (const float*)d_in[11];
    const float* node_b = (const float*)d_in[12];
    const float* dec_w  = (const float*)d_in[13];
    const float* dec_b  = (const float*)d_in[14];
    // d_in[15] = predict_steps == 1 (asserted in reference)

    float* ts  = (float*)d_ws;        // 256*128*32 f32 = 4 MB scratch
    float* out = (float*)d_out;       // 256*128*1*4 f32

    swarm_conv_kernel<<<dim3(4, NA), 256, 0, stream>>>(x, w1, b1, w2, b2, w3, b3, ts);
    swarm_gnn_kernel<<<dim3(NT), 256, 0, stream>>>(ts, x, edge_w, edge_b,
                                                   agg_w, agg_b, node_w, node_b,
                                                   dec_w, dec_b, out);
}
